// AdaptiveSparseWorldBlock_36249523978353
// MI455X (gfx1250) — matrix operand order repaired and checked by reading, not compiled, not run on hardware
//
#include <hip/hip_runtime.h>

// ---------------------------------------------------------------------------
// Types for CDNA5 WMMA
// ---------------------------------------------------------------------------
typedef __attribute__((ext_vector_type(16))) __bf16 v16bf;
typedef __attribute__((ext_vector_type(8)))  __bf16 v8bf;
typedef __attribute__((ext_vector_type(8)))  float  v8f;
typedef __attribute__((vector_size(16)))     int    v4i_raw;

#define D_MODEL 1024
#define L_LAT   4096
#define B_ROWS  4096
#define M_SLOTS 8

enum { EPI_F32 = 0, EPI_BF16 = 1, EPI_SPA = 2, EPI_SPB = 3, EPI_TANH = 4 };

struct GemmP {
    const __bf16* X;    // [M, ldx] row-major activations (bf16)
    const __bf16* W;    // [N, K]   row-major weights (bf16), C = X @ W^T
    const float*  bias; // [N] or nullptr
    void*         out;  // f32 or bf16 per epi, ld = N
    const float*  as;   // a_s for EPI_SPB
    const float*  th;   // thresholds  (sparse epilogues)
    const float*  gn;   // gains
    const float*  mg;   // mult_gate
    int ldx, N, K, epi;
};

// ---------------------------------------------------------------------------
// CDNA5 async global->LDS copy (ASYNCcnt) — builtin if present, else inline asm
// Builtin signature (from probe diagnostic): (v4i AS1*, v4i AS3*, imm, imm)
// ---------------------------------------------------------------------------
__device__ __forceinline__ void async_ld_b128(void* lds_generic, const void* gptr) {
#if __has_builtin(__builtin_amdgcn_global_load_async_to_lds_b128)
    typedef __attribute__((address_space(1))) v4i_raw* gp_t;
    typedef __attribute__((address_space(3))) v4i_raw* lp_t;
    __builtin_amdgcn_global_load_async_to_lds_b128(
        (gp_t)(void*)gptr, (lp_t)lds_generic, 0, 0);
#else
    unsigned loff = (unsigned)(unsigned long long)
        (__attribute__((address_space(3))) char*)lds_generic;
    unsigned long long ga = (unsigned long long)gptr;
    asm volatile("global_load_async_to_lds_b128 %0, %1, off"
                 :: "v"(loff), "v"(ga) : "memory");
#endif
}

__device__ __forceinline__ void wait_async0() {
#if __has_builtin(__builtin_amdgcn_s_wait_asynccnt)
    __builtin_amdgcn_s_wait_asynccnt(0);
#else
    asm volatile("s_wait_asynccnt 0" ::: "memory");
#endif
}

// ---------------------------------------------------------------------------
// WMMA GEMM: block tile 128x128, 8 waves (wave32), wave tile 32x64 (2x4 WMMAs)
// A/B tiles (128 x 32 bf16) staged in LDS via async loads, double-buffered.
// LDS row stride 80B (32 bf16 + 8 pad) keeps 16B alignment, spreads banks.
// A frag (16x32 bf16): lane<16 -> K {0..7,16..23}; lane>=16 -> K {8..15,24..31}
// B frag (32x16 bf16): lane&15 -> col (row of W); same K split as A frag
// ---------------------------------------------------------------------------
#define KSTEP      32
#define LDS_STRIDE 80
#define TILE_BYTES (128 * LDS_STRIDE)   // 10240
#define BUF_BYTES  (2 * TILE_BYTES)     // A + B = 20480

__global__ __launch_bounds__(256) void wmma_gemm(GemmP p) {
    __shared__ __align__(16) char smem[2 * BUF_BYTES];  // 40 KB, double buffered
    const int t    = threadIdx.x;
    const int lane = t & 31;
    const int wid  = t >> 5;
    const int wm = wid & 3;          // 4 waves along M
    const int wn = wid >> 2;         // 2 waves along N
    const int rblk = blockIdx.y * 128;
    const int cblk = blockIdx.x * 128;
    const int l15 = lane & 15;
    const int lhi = (lane >> 4) & 1;

    // Per-thread async-copy assignment: 512 16B chunks per tile, 2 per thread.
    // chunk c: row/col = c>>2, kc = c&3.
    const int c0 = t * 2, c1 = t * 2 + 1;
    const int r0 = c0 >> 2, kc0 = c0 & 3;
    const int r1 = c1 >> 2, kc1 = c1 & 3;
    const __bf16* gA0 = p.X + (size_t)(rblk + r0) * p.ldx + kc0 * 8;
    const __bf16* gA1 = p.X + (size_t)(rblk + r1) * p.ldx + kc1 * 8;
    const __bf16* gB0 = p.W + (size_t)(cblk + r0) * p.K + kc0 * 8;
    const __bf16* gB1 = p.W + (size_t)(cblk + r1) * p.K + kc1 * 8;
    const unsigned lo0 = r0 * LDS_STRIDE + kc0 * 16;
    const unsigned lo1 = r1 * LDS_STRIDE + kc1 * 16;

    auto prefetch = [&](int kt, int buf) {
        const int kk = kt * KSTEP;
        char* s = smem + buf * BUF_BYTES;
        async_ld_b128(s + lo0, gA0 + kk);
        async_ld_b128(s + lo1, gA1 + kk);
        async_ld_b128(s + TILE_BYTES + lo0, gB0 + kk);
        async_ld_b128(s + TILE_BYTES + lo1, gB1 + kk);
    };

    v8f acc[2][4] = {};
    const int KT = p.K / KSTEP;
    prefetch(0, 0);

    for (int kt = 0; kt < KT; ++kt) {
        const int cur = kt & 1;
        wait_async0();       // own async loads for buf[cur] landed in LDS
        __syncthreads();     // all waves' tile data visible; prev compute done
        if (kt + 1 < KT) prefetch(kt + 1, cur ^ 1);

        const char* As = smem + cur * BUF_BYTES;
        const char* Bs = As + TILE_BYTES;
        v16bf a[2], b[4];
#pragma unroll
        for (int i = 0; i < 2; ++i) {
            const int r = wm * 32 + i * 16 + l15;
            const __bf16* ap = (const __bf16*)(As + r * LDS_STRIDE) + lhi * 8;
            union { v16bf v; v8bf h[2]; } u;
            u.h[0] = *(const v8bf*)(ap);
            u.h[1] = *(const v8bf*)(ap + 16);
            a[i] = u.v;
        }
#pragma unroll
        for (int j = 0; j < 4; ++j) {
            const int c = wn * 64 + j * 16 + l15;
            const __bf16* bp = (const __bf16*)(Bs + c * LDS_STRIDE) + lhi * 8;
            union { v16bf v; v8bf h[2]; } u;
            u.h[0] = *(const v8bf*)(bp);
            u.h[1] = *(const v8bf*)(bp + 16);
            b[j] = u.v;
        }
#pragma unroll
        for (int i = 0; i < 2; ++i)
#pragma unroll
            for (int j = 0; j < 4; ++j)
                acc[i][j] = __builtin_amdgcn_wmma_f32_16x16x32_bf16(
                    false, a[i], false, b[j], (short)0, acc[i][j], false, false);
    }

    // Epilogue. C/D layout: VGPR e -> row (e or e+8 by lane-half), col = lane&15.
#pragma unroll
    for (int i = 0; i < 2; ++i) {
#pragma unroll
        for (int j = 0; j < 4; ++j) {
            const int col  = cblk + wn * 64 + j * 16 + l15;
            const int row0 = rblk + wm * 32 + i * 16 + lhi * 8;
            const float bias = p.bias ? p.bias[col] : 0.0f;
            const int coh = col >> 9;  // L/C = 512 (sparse epilogues only; N==4096)
#pragma unroll
            for (int e = 0; e < 8; ++e) {
                const int row = row0 + e;
                float v = acc[i][j][e] + bias;
                const size_t idx = (size_t)row * p.N + col;
                switch (p.epi) {
                case EPI_F32:
                    ((float*)p.out)[idx] = v; break;
                case EPI_BF16:
                    ((__bf16*)p.out)[idx] = (__bf16)v; break;
                case EPI_SPA: {
                    float tt = v - p.th[coh]; tt = tt > 0.f ? tt : 0.f;
                    ((float*)p.out)[idx] = tt * p.gn[coh];
                } break;
                case EPI_SPB: {
                    float tt = v - p.th[coh]; tt = tt > 0.f ? tt : 0.f;
                    tt = tt * p.gn[coh] * p.as[idx] * p.mg[coh];
                    ((__bf16*)p.out)[idx] = (__bf16)tt;
                } break;
                case EPI_TANH:
                    ((__bf16*)p.out)[idx] = (__bf16)tanhf(v); break;
                }
            }
        }
    }
}

// ---------------------------------------------------------------------------
// Elementwise / row kernels
// ---------------------------------------------------------------------------
__device__ __forceinline__ float block_reduce_sum(float v, float* sbuf) {
    const int t = threadIdx.x;
    sbuf[t] = v; __syncthreads();
    for (int s = 128; s > 0; s >>= 1) {
        if (t < s) sbuf[t] += sbuf[t + s];
        __syncthreads();
    }
    float r = sbuf[0]; __syncthreads();
    return r;
}

__global__ void cvt_bf16_kernel(const float* __restrict__ src,
                                __bf16* __restrict__ dst, int n) {
    const int i = blockIdx.x * 1024 + threadIdx.x;
#pragma unroll
    for (int r = 0; r < 4; ++r) {
        const int j = i + r * 256;
        if (j < n) dst[j] = (__bf16)src[j];
    }
}

// one block per row of length 1024
__global__ __launch_bounds__(256) void rmsnorm_bf16_kernel(
    const float* __restrict__ x, const float* __restrict__ w,
    __bf16* __restrict__ out) {
    __shared__ float sbuf[256];
    const int row = blockIdx.x, t = threadIdx.x;
    const float* xr = x + (size_t)row * D_MODEL;
    float v[4], ss = 0.f;
#pragma unroll
    for (int r = 0; r < 4; ++r) { v[r] = xr[t + r * 256]; ss += v[r] * v[r]; }
    const float tot = block_reduce_sum(ss, sbuf);
    const float sc = 1.0f / sqrtf(tot * (1.0f / D_MODEL) + 1e-6f);
    __bf16* orow = out + (size_t)row * D_MODEL;
#pragma unroll
    for (int r = 0; r < 4; ++r)
        orow[t + r * 256] = (__bf16)(v[r] * sc * w[t + r * 256]);
}

// per batch row: logits over M=8 slots, softmax, weighted sum of v
__global__ __launch_bounds__(256) void attn_kernel(
    const float* __restrict__ q, const __bf16* __restrict__ kk,
    const __bf16* __restrict__ vv, float* __restrict__ ctx) {
    __shared__ float sbuf[256];
    __shared__ float lg[M_SLOTS];
    const int b = blockIdx.x, t = threadIdx.x;
    const float* qr = q + (size_t)b * D_MODEL;
    float qv[4];
#pragma unroll
    for (int r = 0; r < 4; ++r) qv[r] = qr[t + r * 256];
    float part[M_SLOTS];
#pragma unroll
    for (int m = 0; m < M_SLOTS; ++m) {
        const __bf16* kr = kk + ((size_t)b * M_SLOTS + m) * D_MODEL;
        float s = 0.f;
#pragma unroll
        for (int r = 0; r < 4; ++r) s += qv[r] * (float)kr[t + r * 256];
        part[m] = s;
    }
    for (int m = 0; m < M_SLOTS; ++m) {
        const float tot = block_reduce_sum(part[m], sbuf);
        if (t == 0) lg[m] = tot * 0.03125f;  // 1/sqrt(1024)
        __syncthreads();
    }
    float mx = lg[0];
    for (int m = 1; m < M_SLOTS; ++m) mx = fmaxf(mx, lg[m]);
    float aw[M_SLOTS], den = 0.f;
    for (int m = 0; m < M_SLOTS; ++m) { aw[m] = __expf(lg[m] - mx); den += aw[m]; }
    const float inv = 1.0f / den;
    float* cr = ctx + (size_t)b * D_MODEL;
#pragma unroll
    for (int r = 0; r < 4; ++r) {
        const int d = t + r * 256;
        float s = 0.f;
#pragma unroll
        for (int m = 0; m < M_SLOTS; ++m)
            s += aw[m] * (float)vv[((size_t)b * M_SLOTS + m) * D_MODEL + d];
        cr[d] = s * inv;
    }
}

// ctx = rmsnorm(context+hidden)*norm_state (bf16); concat = [xn|context|hidden] bf16
__global__ __launch_bounds__(256) void ctx_concat_kernel(
    const float* __restrict__ context, const float* __restrict__ hidden,
    const float* __restrict__ norm_state, const __bf16* __restrict__ xn,
    __bf16* __restrict__ ctxo, __bf16* __restrict__ concat) {
    __shared__ float sbuf[256];
    const int b = blockIdx.x, t = threadIdx.x;
    const float* cr = context + (size_t)b * D_MODEL;
    const float* hr = hidden + (size_t)b * D_MODEL;
    float sv[4], cv[4], hv[4], ss = 0.f;
#pragma unroll
    for (int r = 0; r < 4; ++r) {
        const int d = t + r * 256;
        cv[r] = cr[d]; hv[r] = hr[d]; sv[r] = cv[r] + hv[r]; ss += sv[r] * sv[r];
    }
    const float tot = block_reduce_sum(ss, sbuf);
    const float sc = 1.0f / sqrtf(tot * (1.0f / D_MODEL) + 1e-6f);
    __bf16* co = ctxo + (size_t)b * D_MODEL;
    const __bf16* xr = xn + (size_t)b * D_MODEL;
    __bf16* cc = concat + (size_t)b * (3 * D_MODEL);
#pragma unroll
    for (int r = 0; r < 4; ++r) {
        const int d = t + r * 256;
        co[d] = (__bf16)(sv[r] * sc * norm_state[d]);
        cc[d] = xr[d];
        cc[D_MODEL + d] = (__bf16)cv[r];
        cc[2 * D_MODEL + d] = (__bf16)hv[r];
    }
}

// x_out rmsnorm, GRU combine, next_hidden, normalized slot write + memory shift
__global__ __launch_bounds__(256) void final_kernel(
    const float* __restrict__ x_t, const float* __restrict__ hidden,
    const float* __restrict__ memory, const float* __restrict__ delta,
    const float* __restrict__ gi, const float* __restrict__ gh,
    const float* __restrict__ norm_out, const float* __restrict__ rg,
    const float* __restrict__ wap, const float* __restrict__ flp,
    float* __restrict__ out_x, float* __restrict__ out_h,
    float* __restrict__ out_mem) {
    __shared__ float sbuf[256];
    const int b = blockIdx.x, t = threadIdx.x;
    const float rscale = rg[0], wa = wap[0], fl = flp[0];

    const float* xr = x_t + (size_t)b * D_MODEL;
    const float* dr = delta + (size_t)b * D_MODEL;
    float yv[4], ss = 0.f;
#pragma unroll
    for (int r = 0; r < 4; ++r) {
        const int d = t + r * 256;
        yv[r] = xr[d] + rscale * dr[d]; ss += yv[r] * yv[r];
    }
    const float tot = block_reduce_sum(ss, sbuf);
    const float sc = 1.0f / sqrtf(tot * (1.0f / D_MODEL) + 1e-6f);
    float* oxr = out_x + (size_t)b * D_MODEL;
#pragma unroll
    for (int r = 0; r < 4; ++r) {
        const int d = t + r * 256;
        oxr[d] = yv[r] * sc * norm_out[d];
    }

    const float* gir = gi + (size_t)b * (3 * D_MODEL);
    const float* ghr = gh + (size_t)b * (3 * D_MODEL);
    const float* hrp = hidden + (size_t)b * D_MODEL;
    float nh[4], s2 = 0.f;
#pragma unroll
    for (int r = 0; r < 4; ++r) {
        const int d = t + r * 256;
        const float ir = gir[d], iz = gir[D_MODEL + d], inn = gir[2 * D_MODEL + d];
        const float hrv = ghr[d], hz = ghr[D_MODEL + d], hn = ghr[2 * D_MODEL + d];
        const float rr = 1.0f / (1.0f + __expf(-(ir + hrv)));
        const float zz = 1.0f / (1.0f + __expf(-(iz + hz)));
        const float nn = tanhf(inn + rr * hn);
        const float h = hrp[d];
        const float cand = (1.0f - zz) * nn + zz * h;
        const float v = fl * h + wa * cand;
        nh[r] = v; s2 += v * v;
    }
    const float tot2 = block_reduce_sum(s2, sbuf);
    const float nrm = fmaxf(sqrtf(tot2), 1e-12f);
    const float inv = wa / nrm;
    float* ohr = out_h + (size_t)b * D_MODEL;
    float* omr = out_mem + (size_t)b * (M_SLOTS * D_MODEL);
#pragma unroll
    for (int r = 0; r < 4; ++r) {
        const int d = t + r * 256;
        ohr[d] = nh[r];
        omr[d] = nh[r] * inv;
    }
    const float* mr = memory + (size_t)b * (M_SLOTS * D_MODEL);
    for (int slot = 1; slot < M_SLOTS; ++slot)
#pragma unroll
        for (int r = 0; r < 4; ++r) {
            const int d = t + r * 256;
            omr[slot * D_MODEL + d] = mr[(slot - 1) * D_MODEL + d] * fl;
        }
}

// ---------------------------------------------------------------------------
// Host
// ---------------------------------------------------------------------------
static void launch_gemm(const __bf16* X, int ldx, const __bf16* W, const float* bias,
                        void* out, int M, int N, int K, int epi,
                        const float* as, const float* th, const float* gn,
                        const float* mg, hipStream_t stream) {
    GemmP p;
    p.X = X; p.W = W; p.bias = bias; p.out = out;
    p.as = as; p.th = th; p.gn = gn; p.mg = mg;
    p.ldx = ldx; p.N = N; p.K = K; p.epi = epi;
    dim3 grid(N / 128, M / 128);
    wmma_gemm<<<grid, 256, 0, stream>>>(p);
}

static void launch_cvt(const float* src, __bf16* dst, int n, hipStream_t s) {
    cvt_bf16_kernel<<<n / 1024, 256, 0, s>>>(src, dst, n);
}

extern "C" void kernel_launch(void* const* d_in, const int* in_sizes, int n_in,
                              void* d_out, int out_size, void* d_ws, size_t ws_size,
                              hipStream_t stream) {
    (void)in_sizes; (void)n_in; (void)out_size; (void)ws_size;
    const float* x_t        = (const float*)d_in[0];
    const float* hidden     = (const float*)d_in[1];
    const float* memory     = (const float*)d_in[2];
    const float* norm_in    = (const float*)d_in[3];
    const float* norm_out   = (const float*)d_in[4];
    const float* norm_state = (const float*)d_in[5];
    const float* norm_mem   = (const float*)d_in[6];
    const float* Wq   = (const float*)d_in[7];
    const float* Wk   = (const float*)d_in[8];
    const float* Wv   = (const float*)d_in[9];
    const float* Wa   = (const float*)d_in[10];
    const float* ba   = (const float*)d_in[11];
    const float* Wb   = (const float*)d_in[12];
    const float* bb   = (const float*)d_in[13];
    const float* Wdec = (const float*)d_in[14];
    const float* bdec = (const float*)d_in[15];
    const float* Wtr  = (const float*)d_in[16];
    const float* btr  = (const float*)d_in[17];
    const float* W_ih = (const float*)d_in[18];
    const float* W_hh = (const float*)d_in[19];
    const float* b_ih = (const float*)d_in[20];
    const float* b_hh = (const float*)d_in[21];
    const float* th_a = (const float*)d_in[22];
    const float* th_b = (const float*)d_in[23];
    const float* gn_a = (const float*)d_in[24];
    const float* gn_b = (const float*)d_in[25];
    const float* mgat = (const float*)d_in[26];
    const float* rgat = (const float*)d_in[27];
    const float* walp = (const float*)d_in[28];
    const float* flam = (const float*)d_in[29];

    char* base = (char*)d_ws;
    size_t off = 0;
    auto bump = [&](size_t n) -> char* {
        char* p = base + off;
        off += (n + 255) & ~(size_t)255;
        return p;
    };
    __bf16* wq_bf   = (__bf16*)bump(2097152);
    __bf16* wk_bf   = (__bf16*)bump(2097152);
    __bf16* wv_bf   = (__bf16*)bump(2097152);
    __bf16* wa_bf   = (__bf16*)bump(8388608);
    __bf16* wb_bf   = (__bf16*)bump(8388608);
    __bf16* wdec_bf = (__bf16*)bump(8388608);
    __bf16* wtr_bf  = (__bf16*)bump(6291456);
    __bf16* wih_bf  = (__bf16*)bump(6291456);
    __bf16* whh_bf  = (__bf16*)bump(6291456);
    __bf16* xn_bf     = (__bf16*)bump(8388608);
    __bf16* ctx_bf    = (__bf16*)bump(8388608);
    __bf16* concat_bf = (__bf16*)bump(25165824);
    __bf16* prop_bf   = (__bf16*)bump(8388608);
    __bf16* mult_bf   = (__bf16*)bump(33554432);
    float*  context_f = (float*)bump(16777216);
    char*   qdelta = bump(16777216);   // q f32 (dead after attn) -> delta f32
    char*   regA   = bump(67108864);   // mem_bf 64MB (dead after v GEMM) -> a_s 64MB
    char*   regB   = bump(134217728);  // k|v 128MB (dead after attn) -> gi|gh 96MB
    float*  q_f     = (float*)qdelta;
    float*  delta_f = (float*)qdelta;
    __bf16* mem_bf  = (__bf16*)regA;
    float*  a_s     = (float*)regA;
    __bf16* k_bf    = (__bf16*)regB;
    __bf16* v_bf    = (__bf16*)(regB + 67108864);
    float*  gi      = (float*)regB;
    float*  gh      = (float*)(regB + 50331648);

    float* out_x   = (float*)d_out;
    float* out_h   = out_x + (size_t)B_ROWS * D_MODEL;
    float* out_mem = out_h + (size_t)B_ROWS * D_MODEL;

    // 1) weights -> bf16
    launch_cvt(Wq, wq_bf, 1048576, stream);
    launch_cvt(Wk, wk_bf, 1048576, stream);
    launch_cvt(Wv, wv_bf, 1048576, stream);
    launch_cvt(Wa, wa_bf, 4194304, stream);
    launch_cvt(Wb, wb_bf, 4194304, stream);
    launch_cvt(Wdec, wdec_bf, 4194304, stream);
    launch_cvt(Wtr, wtr_bf, 3145728, stream);
    launch_cvt(W_ih, wih_bf, 3145728, stream);
    launch_cvt(W_hh, whh_bf, 3145728, stream);

    // 2) rmsnorms -> bf16 activations
    rmsnorm_bf16_kernel<<<B_ROWS, 256, 0, stream>>>(x_t, norm_in, xn_bf);
    rmsnorm_bf16_kernel<<<B_ROWS * M_SLOTS, 256, 0, stream>>>(memory, norm_mem, mem_bf);

    // 3) q / k / v GEMMs (WMMA)
    launch_gemm(xn_bf, D_MODEL, wq_bf, nullptr, q_f, B_ROWS, D_MODEL, D_MODEL,
                EPI_F32, nullptr, nullptr, nullptr, nullptr, stream);
    launch_gemm(mem_bf, D_MODEL, wk_bf, nullptr, k_bf, B_ROWS * M_SLOTS, D_MODEL,
                D_MODEL, EPI_BF16, nullptr, nullptr, nullptr, nullptr, stream);
    launch_gemm(mem_bf, D_MODEL, wv_bf, nullptr, v_bf, B_ROWS * M_SLOTS, D_MODEL,
                D_MODEL, EPI_BF16, nullptr, nullptr, nullptr, nullptr, stream);

    // 4) attention over M=8 slots
    attn_kernel<<<B_ROWS, 256, 0, stream>>>(q_f, k_bf, v_bf, context_f);

    // 5) ctx rmsnorm + concat build
    ctx_concat_kernel<<<B_ROWS, 256, 0, stream>>>(context_f, hidden, norm_state,
                                                  xn_bf, ctx_bf, concat_bf);

    // 6) cohort-sparse latents: a_s (f32), mult (bf16, fused a*b gating)
    launch_gemm(xn_bf, D_MODEL, wa_bf, ba, a_s, B_ROWS, L_LAT, D_MODEL,
                EPI_SPA, nullptr, th_a, gn_a, nullptr, stream);
    launch_gemm(ctx_bf, D_MODEL, wb_bf, bb, mult_bf, B_ROWS, L_LAT, D_MODEL,
                EPI_SPB, a_s, th_b, gn_b, mgat, stream);

    // 7) delta = mult @ Wdec^T
    launch_gemm(mult_bf, L_LAT, wdec_bf, bdec, delta_f, B_ROWS, D_MODEL, L_LAT,
                EPI_F32, nullptr, nullptr, nullptr, nullptr, stream);

    // 8) prop = tanh(concat @ Wtr^T + btr) -> bf16
    launch_gemm(concat_bf, 3 * D_MODEL, wtr_bf, btr, prop_bf, B_ROWS, D_MODEL,
                3 * D_MODEL, EPI_TANH, nullptr, nullptr, nullptr, nullptr, stream);

    // 9) gi = prop @ W_ih^T + b_ih; gh = hidden @ W_hh^T + b_hh
    launch_gemm(prop_bf, D_MODEL, wih_bf, b_ih, gi, B_ROWS, 3 * D_MODEL, D_MODEL,
                EPI_F32, nullptr, nullptr, nullptr, nullptr, stream);
    launch_gemm(concat_bf + 2 * D_MODEL, 3 * D_MODEL, whh_bf, b_hh, gh, B_ROWS,
                3 * D_MODEL, D_MODEL, EPI_F32, nullptr, nullptr, nullptr, nullptr,
                stream);

    // 10) x_out rmsnorm + GRU combine + memory write
    final_kernel<<<B_ROWS, 256, 0, stream>>>(x_t, hidden, memory, delta_f, gi, gh,
                                             norm_out, rgat, walp, flam,
                                             out_x, out_h, out_mem);
}
